// DeepSet_59098749993058
// MI455X (gfx1250) — matrix operand order repaired and checked by reading, not compiled, not run on hardware
//
#include <hip/hip_runtime.h>
#include <hip/hip_bf16.h>
#include <stdint.h>

// ---------------------------------------------------------------------------
// CDNA5 (gfx1250) wave32 implementation of the DeepSet GNN forward pass.
//  - all matmuls via v_wmma_f32_16x16x32_f16 (16-node tile per wave)
//  - weights pre-converted once to transposed/padded f16 [n][Kpad] blocks,
//    then DMA'd per-workgroup into LDS with tensor_load_to_lds (TDM)
//  - node state h kept in f16 (HBM-bandwidth-bound problem)
//  - segment sum/max via run-compacted atomics over sorted segment ids
// ---------------------------------------------------------------------------

typedef __attribute__((ext_vector_type(16))) _Float16 v16h;
typedef __attribute__((ext_vector_type(8)))  float    v8f;
typedef __attribute__((ext_vector_type(4)))  unsigned u32x4;
typedef __attribute__((ext_vector_type(8)))  int      i32x8;
typedef __attribute__((ext_vector_type(4)))  int      i32x4;

#define WMMA(a, b, c) __builtin_amdgcn_wmma_f32_16x16x32_f16( \
    false, (a), false, (b), (short)0, (c), false, false)

static __device__ __forceinline__ unsigned fkey(float x) {
  unsigned b = __float_as_uint(x);
  return (b & 0x80000000u) ? ~b : (b | 0x80000000u);   // monotonic float->uint
}
static __device__ __forceinline__ float fdec(unsigned k) {
  unsigned b = (k & 0x80000000u) ? (k & 0x7fffffffu) : ~k;
  return __uint_as_float(b);
}
static __device__ __forceinline__ float prelu(float x, float a) {
  return x >= 0.f ? x : a * x;
}
static __device__ __forceinline__ float sel4(float a, float b, float c, float d, int k) {
  float v = a; v = (k == 1) ? b : v; v = (k == 2) ? c : v; v = (k == 3) ? d : v;
  return v;
}

union V16 { v16h v; uint4 q[2]; };
// 16 halves from two 16B chunks (p0: halves 0..7, p1: halves 8..15)
static __device__ __forceinline__ v16h lds_pair(const _Float16* p0, const _Float16* p1) {
  V16 u;
  u.q[0] = *reinterpret_cast<const uint4*>(p0);
  u.q[1] = *reinterpret_cast<const uint4*>(p1);
  return u.v;
}
static __device__ __forceinline__ v16h lds_single(const _Float16* p0) {
  V16 u;
  u.q[0] = *reinterpret_cast<const uint4*>(p0);
  u.q[1] = make_uint4(0u, 0u, 0u, 0u);
  return u.v;
}

// ---- Tensor Data Mover: 1-D copy global->LDS (D# per ISA 08_async_tensor §8)
// 6-arg builtin form (amdgpu-toolchain clang-23): (u32x4, i32x8, i32x4, i32x4, i32x8, i32)
static __device__ __forceinline__ void tdm_load(unsigned lds_off, const void* gsrc,
                                                unsigned dwords) {
  unsigned long long ga = (unsigned long long)(size_t)gsrc;
  u32x4 g0;
  g0[0] = 1u;                                             // count=1 (valid user D#)
  g0[1] = lds_off;                                        // lds_addr (bytes)
  g0[2] = (unsigned)ga;                                   // global_addr[31:0]
  g0[3] = (unsigned)((ga >> 32) & 0x01FFFFFFu) | 0x80000000u;  // ga[56:32] | type=2
  i32x8 g1;
  g1[0] = (int)(2u << 16);                                // data_size = 4B
  g1[1] = (int)((dwords & 0xFFFFu) << 16);                // tensor_dim0 lo16
  g1[2] = (int)(((dwords >> 16) & 0xFFFFu) | (1u << 16)); // tensor_dim0 hi16 | tensor_dim1=1
  g1[3] = (int)((dwords & 0xFFFFu) << 16);                // tile_dim0
  g1[4] = 1;                                              // tile_dim1 = 1
  g1[5] = (int)dwords;                                    // tensor_dim0_stride lo32
  g1[6] = 0;
  g1[7] = 0;
  i32x4 gz4 = {0, 0, 0, 0};
  i32x8 gz8 = {0, 0, 0, 0, 0, 0, 0, 0};
  __builtin_amdgcn_tensor_load_to_lds(g0, g1, gz4, gz4, gz8, 0);
}

struct InitP { const _Float16* Wt; const float *b1, *b2, *a1, *a2; };
struct HidP  { const _Float16* Wt; const float *b0, *b1, *b2, *a0, *a1, *a2; };

// weight prep: f32 [K][16] -> f16 transposed+padded [16][Kpad]
__global__ __launch_bounds__(256)
void wprep(const float* __restrict__ W, int K, int Kpad, _Float16* __restrict__ out) {
  int i = blockIdx.x * 256 + threadIdx.x;
  if (i >= 16 * Kpad) return;
  int n = i / Kpad, k = i % Kpad;
  out[i] = (k < K) ? (_Float16)W[k * 16 + n] : (_Float16)0.f;
}

// ------------------------- init MLP: fin->16->16 ---------------------------
template <int TYPE>  // 0=cells(F=4), 1=tracks(F=4), 2=subjets(F=3)
__global__ __launch_bounds__(256)
void init_kernel(const float* __restrict__ center, const float* __restrict__ aux,
                 const float* __restrict__ E, const float* __restrict__ globE,
                 const int* __restrict__ seg, int N, InitP P,
                 _Float16* __restrict__ hOut) {
  constexpr int F = (TYPE == 2) ? 3 : 4;
  __shared__ __align__(16) _Float16 stile[8][256];
  __shared__ __align__(16) _Float16 wlds[1024];   // Wt1[16][32] + Wt2[16][32]

  const int w    = threadIdx.x >> 5;
  const int lane = threadIdx.x & 31;
  const int numTiles = (N + 15) >> 4;
  const int tileR = blockIdx.x * 8 + w;
  const bool valid = tileR < numTiles;
  const int tile = valid ? tileR : (numTiles - 1);

  if (threadIdx.x < 32)                       // wave 0 issues the weight DMA
    tdm_load((unsigned)(size_t)&wlds[0], P.Wt, 1024 * 2 / 4);

  const int m  = lane & 15;          // A-row / D-column owned by this lane
  const int hi = (lane >> 4) & 1;    // upper half-wave
  const int n  = m;

  int node  = tile * 16 + m;
  int nodeC = node < N ? node : N - 1;
  float f0 = center[(size_t)nodeC * 2 + 0];
  float f1 = center[(size_t)nodeC * 2 + 1];
  float ge = globE[seg[nodeC]];
  float f2, f3;
  if (TYPE == 0)      { f2 = aux[nodeC] - 2.5f;  f3 = E[nodeC] / ge; }
  else if (TYPE == 1) { f2 = fabsf(aux[nodeC]);  f3 = E[nodeC] / ge; }
  else                { f2 = E[nodeC] / ge;      f3 = 0.f; }

  v16h a;
#pragma unroll
  for (int j = 0; j < 16; ++j) {
    int k = j + (j & 8) + hi * 8;                         // A layout K map
    a[j] = (k < F) ? (_Float16)sel4(f0, f1, f2, f3, k) : (_Float16)0.f;
  }

  if (threadIdx.x < 32) __builtin_amdgcn_s_wait_tensorcnt(0);
  __syncthreads();                                        // weights now in LDS

  // ---- layer 1 ----
  v16h bW = lds_pair(&wlds[n * 32 + hi * 8], &wlds[n * 32 + hi * 8 + 16]);
  v8f c = {0.f, 0.f, 0.f, 0.f, 0.f, 0.f, 0.f, 0.f};
  c = WMMA(a, bW, c);
  float bias = P.b1[n], al = P.a1[0];
#pragma unroll
  for (int j = 0; j < 8; ++j)
    stile[w][(j + hi * 8) * 16 + n] = (_Float16)prelu(c[j] + bias, al);
  asm volatile("s_wait_dscnt 0" ::: "memory");

  // ---- layer 2 ----
  a  = lds_single(&stile[w][m * 16 + hi * 8]);
  bW = lds_pair(&wlds[512 + n * 32 + hi * 8], &wlds[512 + n * 32 + hi * 8 + 16]);
  v8f c2 = {0.f, 0.f, 0.f, 0.f, 0.f, 0.f, 0.f, 0.f};
  c2 = WMMA(a, bW, c2);
  bias = P.b2[n]; al = P.a2[0];
  asm volatile("s_wait_dscnt 0" ::: "memory");            // WAR on stile
#pragma unroll
  for (int j = 0; j < 8; ++j)
    stile[w][(j + hi * 8) * 16 + n] = (_Float16)prelu(c2[j] + bias, al);
  asm volatile("s_wait_dscnt 0" ::: "memory");

  int r  = lane >> 1;
  int nd = tile * 16 + r;
  if (valid && nd < N) {
    *reinterpret_cast<uint4*>(&hOut[(size_t)nd * 16 + (lane & 1) * 8]) =
        *reinterpret_cast<const uint4*>(&stile[w][r * 16 + (lane & 1) * 8]);
  }
}

// -------------------- hid MLP: (F+64)->16->16->16 --------------------------
template <int TYPE>
__global__ __launch_bounds__(256)
void hid_kernel(const float* __restrict__ center, const float* __restrict__ aux,
                const float* __restrict__ E, const float* __restrict__ globE,
                const int* __restrict__ seg, int N,
                const float* __restrict__ sums, const unsigned* __restrict__ mkey,
                const unsigned* __restrict__ cnt, HidP P,
                _Float16* __restrict__ h) {
  constexpr int F = (TYPE == 2) ? 3 : 4;
  __shared__ __align__(16) _Float16 sfeat[8][16 * 96];
  __shared__ __align__(16) _Float16 stile[8][256];
  __shared__ __align__(16) _Float16 wlds[2560];  // Wt0[16][96] Wt1[16][32] Wt2[16][32]

  const int w    = threadIdx.x >> 5;
  const int lane = threadIdx.x & 31;
  const int numTiles = (N + 15) >> 4;
  const int tileR = blockIdx.x * 8 + w;
  const bool valid = tileR < numTiles;
  const int tile = valid ? tileR : (numTiles - 1);

  if (threadIdx.x < 32)                       // wave 0 issues the weight DMA
    tdm_load((unsigned)(size_t)&wlds[0], P.Wt, 2560 * 2 / 4);

  const int m  = lane & 15;
  const int hi = (lane >> 4) & 1;
  const int n  = m;

  int node  = tile * 16 + m;
  int nodeC = node < N ? node : N - 1;
  int sg    = seg[nodeC];
  float ge  = globE[sg];
  float f0 = center[(size_t)nodeC * 2 + 0];
  float f1 = center[(size_t)nodeC * 2 + 1];
  float f2, f3;
  if (TYPE == 0)      { f2 = aux[nodeC] - 2.5f;  f3 = E[nodeC] / ge; }
  else if (TYPE == 1) { f2 = fabsf(aux[nodeC]);  f3 = E[nodeC] / ge; }
  else                { f2 = E[nodeC] / ge;      f3 = 0.f; }

  const float*    srow = sums + (size_t)sg * 16;
  const unsigned* krow = mkey + (size_t)sg * 16;
  float invc = 1.f / (float)cnt[sg];
  const _Float16* hrow = h + (size_t)nodeC * 16;

  // stage concat([inf, sum, mean, max, h], pad to 96) into LDS (f16)
  int base = hi * 48;
  for (int t = 0; t < 48; ++t) {
    int k = base + t;
    float v;
    if (k < F)             v = sel4(f0, f1, f2, f3, k);
    else if (k < F + 16)   v = srow[k - F];
    else if (k < F + 32)   v = srow[k - F - 16] * invc;
    else if (k < F + 48)   v = fdec(krow[k - F - 32]);
    else if (k < F + 64)   v = (float)hrow[k - F - 48];
    else                   v = 0.f;
    sfeat[w][m * 96 + k] = (_Float16)v;
  }
  asm volatile("s_wait_dscnt 0" ::: "memory");

  if (threadIdx.x < 32) __builtin_amdgcn_s_wait_tensorcnt(0);
  __syncthreads();                                        // weights now in LDS

  // ---- layer 0: [16x96(pad)] @ [96x16] = 3 chained WMMAs ----
  v8f acc = {0.f, 0.f, 0.f, 0.f, 0.f, 0.f, 0.f, 0.f};
#pragma unroll
  for (int cch = 0; cch < 3; ++cch) {
    int kb = cch * 32;
    const _Float16* ab = &sfeat[w][m * 96 + kb + hi * 8];
    v16h a  = lds_pair(ab, ab + 16);
    const _Float16* wb = &wlds[n * 96 + kb + hi * 8];
    v16h bW = lds_pair(wb, wb + 16);
    acc = WMMA(a, bW, acc);
  }
  float bias = P.b0[n], al = P.a0[0];
#pragma unroll
  for (int j = 0; j < 8; ++j)
    stile[w][(j + hi * 8) * 16 + n] = (_Float16)prelu(acc[j] + bias, al);
  asm volatile("s_wait_dscnt 0" ::: "memory");

  // ---- layers 1 and 2: [16x16] @ [16x16] ----
  const float* bs[2]  = {P.b1, P.b2};
  const float* as_[2] = {P.a1, P.a2};
#pragma unroll
  for (int L = 0; L < 2; ++L) {
    v16h a = lds_single(&stile[w][m * 16 + hi * 8]);
    const _Float16* wb = &wlds[1536 + L * 512 + n * 32 + hi * 8];
    v16h bW = lds_pair(wb, wb + 16);
    v8f cc = {0.f, 0.f, 0.f, 0.f, 0.f, 0.f, 0.f, 0.f};
    cc = WMMA(a, bW, cc);
    float bb = bs[L][n], aa = as_[L][0];
    asm volatile("s_wait_dscnt 0" ::: "memory");          // WAR on stile
#pragma unroll
    for (int j = 0; j < 8; ++j)
      stile[w][(j + hi * 8) * 16 + n] = (_Float16)prelu(cc[j] + bb, aa);
    asm volatile("s_wait_dscnt 0" ::: "memory");
  }

  int r  = lane >> 1;
  int nd = tile * 16 + r;
  if (valid && nd < N) {
    *reinterpret_cast<uint4*>(&h[(size_t)nd * 16 + (lane & 1) * 8]) =
        *reinterpret_cast<const uint4*>(&stile[w][r * 16 + (lane & 1) * 8]);
  }
}

// ------------------------- segment reductions ------------------------------
__global__ __launch_bounds__(256)
void zero_u32(unsigned* p, int n) {
  int i = blockIdx.x * 256 + threadIdx.x;
  if (i < n) p[i] = 0u;
}
__global__ __launch_bounds__(256)
void stats_zero(float* s, unsigned* mk, int n) {
  int i = blockIdx.x * 256 + threadIdx.x;
  if (i < n) { s[i] = 0.f; mk[i] = fkey(-3.402823466e38f); }
}
__global__ __launch_bounds__(256)
void count_kernel(const int* __restrict__ seg, int N, unsigned* cnt) {
  int i = blockIdx.x * 256 + threadIdx.x;
  if (i < N) atomicAdd(&cnt[seg[i]], 1u);
}
// sorted segs: run-compacted sum+max, ~1 atomic pair per (run, feature)
__global__ __launch_bounds__(256)
void stats_kernel(const _Float16* __restrict__ h, const int* __restrict__ seg,
                  int N, float* __restrict__ sums, unsigned* __restrict__ mkey) {
  int t = blockIdx.x * 256 + threadIdx.x;
  int f = t & 15;
  int n0 = (t >> 4) * 64;
  if (n0 >= N) return;
  int n1 = n0 + 64; if (n1 > N) n1 = N;
  int cur = seg[n0];
  float acc = 0.f, mx = -3.402823466e38f;
  for (int nd = n0; nd < n1; ++nd) {
    int s = seg[nd];
    if (s != cur) {
      atomicAdd(&sums[(size_t)cur * 16 + f], acc);
      atomicMax(&mkey[(size_t)cur * 16 + f], fkey(mx));
      cur = s; acc = 0.f; mx = -3.402823466e38f;
    }
    float v = (float)h[(size_t)nd * 16 + f];
    acc += v; if (v > mx) mx = v;
  }
  atomicAdd(&sums[(size_t)cur * 16 + f], acc);
  atomicMax(&mkey[(size_t)cur * 16 + f], fkey(mx));
}

// ----------------------------- classifier ----------------------------------
__global__ __launch_bounds__(256)
void clf_kernel(const float* __restrict__ sum_p, const unsigned* __restrict__ cnt_p,
                const float* __restrict__ sum_o, const unsigned* __restrict__ cnt_o,
                const float* __restrict__ sum_t, const unsigned* __restrict__ cnt_t,
                int B,
                const float* W1, const float* b1, const float* a1p,
                const float* W2, const float* b2, const float* a2p,
                const float* W3, const float* b3, const float* a3p,
                const float* W4, const float* b4, float* __restrict__ out) {
  int b = blockIdx.x * 256 + threadIdx.x;
  if (b >= B) return;
  float x[48];
  float ip = 1.f / (float)cnt_p[b];
  float io = 1.f / (float)cnt_o[b];
  float it = 1.f / (float)cnt_t[b];
#pragma unroll
  for (int f = 0; f < 16; ++f) {
    x[f]      = sum_p[(size_t)b * 16 + f] * ip;
    x[16 + f] = sum_o[(size_t)b * 16 + f] * io;
    x[32 + f] = sum_t[(size_t)b * 16 + f] * it;
  }
  float a1 = a1p[0], a2 = a2p[0], a3 = a3p[0];
  float y[48];
  for (int j = 0; j < 48; ++j) {
    float s = b1[j];
#pragma unroll
    for (int i = 0; i < 48; ++i) s += x[i] * W1[i * 48 + j];
    y[j] = prelu(s, a1);
  }
  float z[32];
  for (int j = 0; j < 32; ++j) {
    float s = b2[j];
#pragma unroll
    for (int i = 0; i < 48; ++i) s += y[i] * W2[i * 32 + j];
    z[j] = prelu(s, a2);
  }
  float u[16];
  for (int j = 0; j < 16; ++j) {
    float s = b3[j];
#pragma unroll
    for (int i = 0; i < 32; ++i) s += z[i] * W3[i * 16 + j];
    u[j] = prelu(s, a3);
  }
  float r = b4[0];
#pragma unroll
  for (int i = 0; i < 16; ++i) r += u[i] * W4[i];
  out[b] = r;
}

// ------------------------------- launch ------------------------------------
extern "C" void kernel_launch(void* const* d_in, const int* in_sizes, int n_in,
                              void* d_out, int out_size, void* d_ws, size_t ws_size,
                              hipStream_t stream) {
  const int Np = in_sizes[0] / 2, Nt = in_sizes[3] / 2, No = in_sizes[6] / 2;
  const int B  = in_sizes[8];

  const float* cells_center  = (const float*)d_in[0];
  const float* cells_type    = (const float*)d_in[1];
  const float* cells_E       = (const float*)d_in[2];
  const float* tracks_center = (const float*)d_in[3];
  const float* tracks_D0     = (const float*)d_in[4];
  const float* tracks_E      = (const float*)d_in[5];
  const float* subj_center   = (const float*)d_in[6];
  const float* subj_E        = (const float*)d_in[7];
  const float* glob_E        = (const float*)d_in[8];
  const int*   seg_p         = (const int*)d_in[9];
  const int*   seg_t         = (const int*)d_in[10];
  const int*   seg_o         = (const int*)d_in[11];

  // params = jax tree-flatten (sorted dict keys) => 110 leaves appended last
  const int PB = n_in - 110;
  auto pf = [&](int i) { return (const float*)d_in[i]; };
  const int CLF = PB, HID_O = PB + 11, HID_P = PB + 38, HID_T = PB + 65,
            INIT_O = PB + 92, INIT_P = PB + 98, INIT_T = PB + 104;

  // workspace carve: h (f16) + stats + staged f16 weights
  char* wsb = (char*)d_ws; size_t off = 0;
  auto carve = [&](size_t bytes) -> void* {
    void* p = wsb + off; off = (off + bytes + 255) & ~(size_t)255; return p;
  };
  _Float16* h_p = (_Float16*)carve((size_t)Np * 16 * 2);
  _Float16* h_t = (_Float16*)carve((size_t)Nt * 16 * 2);
  _Float16* h_o = (_Float16*)carve((size_t)No * 16 * 2);
  float*    sum_p = (float*)carve((size_t)B * 16 * 4);
  float*    sum_t = (float*)carve((size_t)B * 16 * 4);
  float*    sum_o = (float*)carve((size_t)B * 16 * 4);
  unsigned* mk_p  = (unsigned*)carve((size_t)B * 16 * 4);
  unsigned* mk_t  = (unsigned*)carve((size_t)B * 16 * 4);
  unsigned* mk_o  = (unsigned*)carve((size_t)B * 16 * 4);
  unsigned* cnt_p = (unsigned*)carve((size_t)B * 4);
  unsigned* cnt_t = (unsigned*)carve((size_t)B * 4);
  unsigned* cnt_o = (unsigned*)carve((size_t)B * 4);
  _Float16* wIni[3];
  for (int t = 0; t < 3; ++t) wIni[t] = (_Float16*)carve(1024 * 2);
  _Float16* wHid[3][3];
  for (int t = 0; t < 3; ++t)
    for (int m2 = 0; m2 < 3; ++m2) wHid[t][m2] = (_Float16*)carve(2560 * 2);
  (void)ws_size; (void)out_size;

  auto tgrid = [&](int N) { return dim3((unsigned)(((N + 15) / 16 + 7) / 8)); };
  auto sgrid = [&](int N) { return dim3((unsigned)((((N + 63) / 64) * 16 + 255) / 256)); };
  auto lgrid = [&](int N) { return dim3((unsigned)((N + 255) / 256)); };
  auto wpg   = [&](int Kpad) { return dim3((unsigned)((16 * Kpad + 255) / 256)); };

  // ---- stage all weights as transposed/padded f16 blocks ----
  const int initBase[3] = {INIT_P, INIT_T, INIT_O};
  const int hidBase[3]  = {HID_P, HID_T, HID_O};
  const int Fin[3]      = {4, 4, 3};
  for (int t = 0; t < 3; ++t) {
    int q = initBase[t];   // [a1,a2,l1.W,l1.b,l2.W,l2.b]
    wprep<<<wpg(32), 256, 0, stream>>>(pf(q + 2), Fin[t], 32, wIni[t]);
    wprep<<<wpg(32), 256, 0, stream>>>(pf(q + 4), 16, 32, wIni[t] + 512);
    for (int m2 = 0; m2 < 3; ++m2) {
      int r = hidBase[t] + 9 * m2;   // [a0,a1,a2,l0.W,l0.b,l1.W,l1.b,l2.W,l2.b]
      wprep<<<wpg(96), 256, 0, stream>>>(pf(r + 3), Fin[t] + 64, 96, wHid[t][m2]);
      wprep<<<wpg(32), 256, 0, stream>>>(pf(r + 5), 16, 32, wHid[t][m2] + 1536);
      wprep<<<wpg(32), 256, 0, stream>>>(pf(r + 7), 16, 32, wHid[t][m2] + 2048);
    }
  }
  auto initP = [&](int t) {
    int q = initBase[t];
    InitP P; P.Wt = wIni[t]; P.b1 = pf(q + 3); P.b2 = pf(q + 5);
    P.a1 = pf(q + 0); P.a2 = pf(q + 1); return P;
  };
  auto hidP = [&](int t, int m2) {
    int q = hidBase[t] + 9 * m2;
    HidP P; P.Wt = wHid[t][m2]; P.b0 = pf(q + 4); P.b1 = pf(q + 6); P.b2 = pf(q + 8);
    P.a0 = pf(q + 0); P.a1 = pf(q + 1); P.a2 = pf(q + 2); return P;
  };

  // segment counts (constant across modules)
  zero_u32<<<lgrid(B), 256, 0, stream>>>(cnt_p, B);
  zero_u32<<<lgrid(B), 256, 0, stream>>>(cnt_t, B);
  zero_u32<<<lgrid(B), 256, 0, stream>>>(cnt_o, B);
  count_kernel<<<lgrid(Np), 256, 0, stream>>>(seg_p, Np, cnt_p);
  count_kernel<<<lgrid(Nt), 256, 0, stream>>>(seg_t, Nt, cnt_t);
  count_kernel<<<lgrid(No), 256, 0, stream>>>(seg_o, No, cnt_o);

  // init MLPs
  init_kernel<0><<<tgrid(Np), 256, 0, stream>>>(cells_center, cells_type, cells_E,
                                                glob_E, seg_p, Np, initP(0), h_p);
  init_kernel<1><<<tgrid(Nt), 256, 0, stream>>>(tracks_center, tracks_D0, tracks_E,
                                                glob_E, seg_t, Nt, initP(1), h_t);
  init_kernel<2><<<tgrid(No), 256, 0, stream>>>(subj_center, nullptr, subj_E,
                                                glob_E, seg_o, No, initP(2), h_o);

  for (int mod = 0; mod < 3; ++mod) {
    stats_zero<<<lgrid(B * 16), 256, 0, stream>>>(sum_p, mk_p, B * 16);
    stats_kernel<<<sgrid(Np), 256, 0, stream>>>(h_p, seg_p, Np, sum_p, mk_p);
    hid_kernel<0><<<tgrid(Np), 256, 0, stream>>>(cells_center, cells_type, cells_E,
        glob_E, seg_p, Np, sum_p, mk_p, cnt_p, hidP(0, mod), h_p);

    stats_zero<<<lgrid(B * 16), 256, 0, stream>>>(sum_t, mk_t, B * 16);
    stats_kernel<<<sgrid(Nt), 256, 0, stream>>>(h_t, seg_t, Nt, sum_t, mk_t);
    hid_kernel<1><<<tgrid(Nt), 256, 0, stream>>>(tracks_center, tracks_D0, tracks_E,
        glob_E, seg_t, Nt, sum_t, mk_t, cnt_t, hidP(1, mod), h_t);

    stats_zero<<<lgrid(B * 16), 256, 0, stream>>>(sum_o, mk_o, B * 16);
    stats_kernel<<<sgrid(No), 256, 0, stream>>>(h_o, seg_o, No, sum_o, mk_o);
    hid_kernel<2><<<tgrid(No), 256, 0, stream>>>(subj_center, nullptr, subj_E,
        glob_E, seg_o, No, sum_o, mk_o, cnt_o, hidP(2, mod), h_o);
  }

  // final per-graph means
  stats_zero<<<lgrid(B * 16), 256, 0, stream>>>(sum_p, mk_p, B * 16);
  stats_kernel<<<sgrid(Np), 256, 0, stream>>>(h_p, seg_p, Np, sum_p, mk_p);
  stats_zero<<<lgrid(B * 16), 256, 0, stream>>>(sum_t, mk_t, B * 16);
  stats_kernel<<<sgrid(Nt), 256, 0, stream>>>(h_t, seg_t, Nt, sum_t, mk_t);
  stats_zero<<<lgrid(B * 16), 256, 0, stream>>>(sum_o, mk_o, B * 16);
  stats_kernel<<<sgrid(No), 256, 0, stream>>>(h_o, seg_o, No, sum_o, mk_o);

  // classifier: x = [mean_p, mean_o, mean_t]; 48->48->32->16->1
  clf_kernel<<<lgrid(B), 256, 0, stream>>>(
      sum_p, cnt_p, sum_o, cnt_o, sum_t, cnt_t, B,
      pf(CLF + 3), pf(CLF + 4), pf(CLF + 0),
      pf(CLF + 5), pf(CLF + 6), pf(CLF + 1),
      pf(CLF + 7), pf(CLF + 8), pf(CLF + 2),
      pf(CLF + 9), pf(CLF + 10), (float*)d_out);
}
